// SmarterClassifierHead_33784212750969
// MI455X (gfx1250) — matrix operand order repaired and checked by reading, not compiled, not run on hardware
//
#include <hip/hip_runtime.h>
#include <hip/hip_bf16.h>

// ---------------------------------------------------------------------------
// SmarterClassifierHead for MI455X (gfx1250, wave32, WMMA).
// All heavy GEMMs run through v_wmma_f32_16x16x32_bf16 (K=32 per issue).
// Wave tile is 64x64 (16 WMMA tiles, 128 acc VGPRs): 16 b128 loads feed
// 16 WMMAs per K=32 step -> 32 FLOP per L1 byte, halving operand traffic
// vs a 32x32 tile. f32 -> bf16 conversion is a one-time ~350MB memory pass
// (~20us at 23.3 TB/s), negligible vs ~1.26 TFLOP of matrix compute.
// ---------------------------------------------------------------------------

typedef __attribute__((ext_vector_type(16))) __bf16 bf16x16;
typedef __attribute__((ext_vector_type(8)))  __bf16 bf16x8;
typedef __attribute__((ext_vector_type(8)))  float  f32x8;

#define IN_DIM   2048
#define OUT_DIM  1000
#define OUT_PAD  1024
#define BATCH    8192
#define RG_DIM   512
#define LN_EPS   1e-5f

union FragA {            // 16x32 bf16 A fragment (contiguous K per lane half):
  bf16x16 v;             // lanes 0-15: row=lane, elems 0..15 = K[0..15]
  struct { bf16x8 lo, hi; } h; // lanes 16-31: row=lane-16, K[16..31]
};

// --------------------------- activations -----------------------------------
__device__ __forceinline__ float act_silu(float v) { return v / (1.f + __expf(-v)); }
__device__ __forceinline__ float act_gelu(float v) { return 0.5f * v * (1.f + erff(v * 0.70710678118654752f)); }
__device__ __forceinline__ float act_mish(float v) {
  float sp = (v > 20.f) ? v : log1pf(__expf(v));   // softplus
  return v * tanhf(sp);
}
__device__ __forceinline__ float act_relu(float v) { return fmaxf(v, 0.f); }

// --------------------------- prep: LN + bf16 casts --------------------------
__global__ __launch_bounds__(256)
void prep_x_kernel(const float* __restrict__ x,
                   const float* __restrict__ ln_g, const float* __restrict__ ln_b,
                   __bf16* __restrict__ xb, __bf16* __restrict__ hb,
                   __bf16* __restrict__ shb) {
  __shared__ float red[256];
  const int row = blockIdx.x;
  const float* xr = x + (size_t)row * IN_DIM;
  float s = 0.f, s2 = 0.f;
  for (int i = threadIdx.x; i < IN_DIM; i += 256) {
    float v = xr[i]; s += v; s2 += v * v;
  }
  red[threadIdx.x] = s; __syncthreads();
  for (int st = 128; st > 0; st >>= 1) {
    if (threadIdx.x < st) red[threadIdx.x] += red[threadIdx.x + st];
    __syncthreads();
  }
  const float mean = red[0] * (1.f / IN_DIM);
  __syncthreads();
  red[threadIdx.x] = s2; __syncthreads();
  for (int st = 128; st > 0; st >>= 1) {
    if (threadIdx.x < st) red[threadIdx.x] += red[threadIdx.x + st];
    __syncthreads();
  }
  const float var  = red[0] * (1.f / IN_DIM) - mean * mean;  // population var (jnp.var)
  const float rstd = rsqrtf(var + LN_EPS);
  for (int i = threadIdx.x; i < IN_DIM; i += 256) {
    float v = xr[i];
    float h = (v - mean) * rstd * ln_g[i] + ln_b[i];
    size_t o = (size_t)row * IN_DIM + i;
    xb[o]  = (__bf16)v;
    hb[o]  = (__bf16)h;
    shb[o] = (__bf16)act_silu(h);
  }
}

// --------------------------- weight convert (+N padding) --------------------
__global__ __launch_bounds__(256)
void convert_pad_kernel(const float* __restrict__ src, __bf16* __restrict__ dst,
                        long long srcRows, long long cols, long long dstRows) {
  long long idx = (long long)blockIdx.x * 256 + threadIdx.x;
  long long tot = dstRows * cols;
  if (idx >= tot) return;
  long long r = idx / cols;
  dst[idx] = (r < srcRows) ? (__bf16)src[idx] : (__bf16)0.f;
}

// --------------------------- routers: fold softmax weights ------------------
__global__ __launch_bounds__(256)
void router_kernel(const float* __restrict__ x, const __bf16* __restrict__ hb,
                   const float* __restrict__ rw0, const float* __restrict__ rb0,
                   const float* __restrict__ rw1, const float* __restrict__ rb1,
                   const float* __restrict__ rw2, const float* __restrict__ rb2,
                   const float* __restrict__ rw3, const float* __restrict__ rb3,
                   const float* __restrict__ domw, const float* __restrict__ domb,
                   const float* __restrict__ scales, const float* __restrict__ zetaP,
                   float* __restrict__ cOut,    // [4][BATCH] branch coefficients
                   float* __restrict__ dOut) {  // [4][BATCH] zeta*dom softmax
  __shared__ float red[256];
  __shared__ float res[18];
  const int row = blockIdx.x;
  const float*  xr = x  + (size_t)row * IN_DIM;
  const __bf16* hr = hb + (size_t)row * IN_DIM;
  const float* rws[4] = {rw0, rw1, rw2, rw3};

  float part[18];
#pragma unroll
  for (int p = 0; p < 18; ++p) part[p] = 0.f;

  for (int i = threadIdx.x; i < IN_DIM; i += 256) {
    float xv = xr[i];
    float hv = (float)hr[i];
    int p = 0;
#pragma unroll
    for (int ri = 0; ri < 4; ++ri)
#pragma unroll
      for (int j = 0; j < ri + 2; ++j)
        part[p++] += xv * rws[ri][(size_t)j * IN_DIM + i];
#pragma unroll
    for (int k = 0; k < 4; ++k)
      part[14 + k] += hv * domw[(size_t)k * IN_DIM + i];
  }
  for (int p = 0; p < 18; ++p) {
    red[threadIdx.x] = part[p]; __syncthreads();
    for (int st = 128; st > 0; st >>= 1) {
      if (threadIdx.x < st) red[threadIdx.x] += red[threadIdx.x + st];
      __syncthreads();
    }
    if (threadIdx.x == 0) res[p] = red[0];
    __syncthreads();
  }
  if (threadIdx.x == 0) {
    const float* rbs[4] = {rb0, rb1, rb2, rb3};
    float c[4] = {0.f, 0.f, 0.f, 0.f};
    int off = 0;
    for (int ri = 0; ri < 4; ++ri) {
      const int n = ri + 2;
      float l[5], mx = -1e30f;
      for (int j = 0; j < n; ++j) { l[j] = res[off + j] + rbs[ri][j]; mx = fmaxf(mx, l[j]); }
      float den = 0.f;
      for (int j = 0; j < n; ++j) { l[j] = __expf(l[j] - mx); den += l[j]; }
      // only routes j <= ri are consumed by the reference
      for (int j = 0; j <= ri; ++j) c[j] += scales[ri] * l[j] / den;
      off += n;
    }
    for (int j = 0; j < 4; ++j) cOut[(size_t)j * BATCH + row] = c[j];

    float zeta = *zetaP;
    float dl[4], mx = -1e30f;
    for (int k = 0; k < 4; ++k) { dl[k] = res[14 + k] + domb[k]; mx = fmaxf(mx, dl[k]); }
    float den = 0.f;
    for (int k = 0; k < 4; ++k) { dl[k] = __expf(dl[k] - mx); den += dl[k]; }
    for (int k = 0; k < 4; ++k) dOut[(size_t)k * BATCH + row] = zeta * dl[k] / den;
  }
}

// --------------------------- WMMA GEMM --------------------------------------
// C[M,N] = A[M,K] * Wt[N,K]^T   (A, Wt bf16 row-major; acc f32)
// Block tile 128x256: 8 waves in a 2(M) x 4(N) grid, each wave owns a 64x64
// register tile = 4x4 WMMA fragments (128 acc VGPRs). Per K=32 step:
// 16 b128 loads -> 16 v_wmma_f32_16x16x32_bf16 (1.0 loads/wmma).
// MODE 0: out = acc + bias[col]                  (f32 store, init)
// MODE 1: out += rowscale[row] * acc             (f32 accumulate)
// MODE 2: out += (*scaleP) * tanh(acc+bias[col]) (f32 accumulate)
// MODE 3: mid  = ACTN(acc) as bf16               (intermediate store)
template <int MODE, int ACTN>
__global__ __launch_bounds__(256)
void gemm_bf16_wmma(const __bf16* __restrict__ A, int lda,
                    const __bf16* __restrict__ Bw, int ldb, int K,
                    float* __restrict__ outF, __bf16* __restrict__ outB,
                    int ldo, int Nreal,
                    const float* __restrict__ bias,
                    const float* __restrict__ rowscale,
                    const float* __restrict__ scaleP) {
  const int lane  = threadIdx.x & 31;
  const int wave  = threadIdx.x >> 5;
  const int wm    = wave >> 2;          // 0..1
  const int wn    = wave & 3;           // 0..3
  const int mBase = blockIdx.x * 128 + wm * 64;
  const int nBase = blockIdx.y * 256 + wn * 64;
  const int r15   = lane & 15;
  const int hiHalf = lane >> 4;                 // 0 for lanes 0-15, 1 for 16-31
  const int kA = hiHalf << 4;                   // A: K sub-block 0 or 16
  const int kB = hiHalf << 4;                   // B: K sub-block 0 or 16

  const __bf16* aP[4];
  const __bf16* bP[4];
#pragma unroll
  for (int i = 0; i < 4; ++i) {
    aP[i] = A  + (size_t)(mBase + i * 16 + r15) * lda + kA;
    bP[i] = Bw + (size_t)(nBase + i * 16 + r15) * ldb + kB;
  }

  f32x8 acc[4][4];
#pragma unroll
  for (int mf = 0; mf < 4; ++mf)
#pragma unroll
    for (int nf = 0; nf < 4; ++nf)
      acc[mf][nf] = f32x8{};

#pragma unroll 1
  for (int k = 0; k < K; k += 32) {
    FragA  a[4];
    bf16x16 b[4];
#pragma unroll
    for (int i = 0; i < 4; ++i) {
      a[i].h.lo = *(const bf16x8*)(aP[i] + k);
      a[i].h.hi = *(const bf16x8*)(aP[i] + k + 8);
    }
#pragma unroll
    for (int i = 0; i < 4; ++i)
      b[i] = *(const bf16x16*)(bP[i] + k);   // B 32x16: lane holds 16 contiguous K
#pragma unroll
    for (int mf = 0; mf < 4; ++mf)
#pragma unroll
      for (int nf = 0; nf < 4; ++nf)
        acc[mf][nf] = __builtin_amdgcn_wmma_f32_16x16x32_bf16(
            false, a[mf].v, false, b[nf], (short)0, acc[mf][nf], false, false);
  }

  const int rowOff = hiHalf << 3;      // C layout: VGPR r -> M=r (+8 upper lanes)
  const float sc = (MODE == 2) ? *scaleP : 0.f;

#pragma unroll
  for (int mf = 0; mf < 4; ++mf) {
#pragma unroll
    for (int nf = 0; nf < 4; ++nf) {
      const int col = nBase + nf * 16 + r15;
      if (col >= Nreal) continue;      // N padded to 64 in converted weights
      const int rb = mBase + mf * 16 + rowOff;
      f32x8 a = acc[mf][nf];
#pragma unroll
      for (int r = 0; r < 8; ++r) {
        const size_t o = (size_t)(rb + r) * ldo + col;
        const float v = a[r];
        if (MODE == 0) {
          outF[o] = v + bias[col];
        } else if (MODE == 1) {
          outF[o] += rowscale[rb + r] * v;
        } else if (MODE == 2) {
          outF[o] += sc * tanhf(v + bias[col]);
        } else {
          float av;
          if (ACTN == 0)      av = act_silu(v);
          else if (ACTN == 1) av = act_gelu(v);
          else if (ACTN == 2) av = act_mish(v);
          else                av = act_relu(v);
          outB[o] = (__bf16)av;
        }
      }
    }
  }
}

// --------------------------- launch ------------------------------------------
extern "C" void kernel_launch(void* const* d_in, const int* in_sizes, int n_in,
                              void* d_out, int out_size, void* d_ws, size_t ws_size,
                              hipStream_t stream) {
  const float* x    = (const float*)d_in[0];
  const float* W    = (const float*)d_in[1];
  const float* bv   = (const float*)d_in[2];
  const float *up_w[4], *down_w[4], *r_w[4], *r_b[4];
  for (int i = 0; i < 4; ++i) {
    up_w[i]   = (const float*)d_in[3 + 4 * i];
    down_w[i] = (const float*)d_in[4 + 4 * i];
    r_w[i]    = (const float*)d_in[5 + 4 * i];
    r_b[i]    = (const float*)d_in[6 + 4 * i];
  }
  const float* scales  = (const float*)d_in[19];
  const float* ln_g    = (const float*)d_in[20];
  const float* ln_b    = (const float*)d_in[21];
  const float* dom_rw  = (const float*)d_in[22];
  const float* dom_rb  = (const float*)d_in[23];
  const float* dom_ew  = (const float*)d_in[24];
  const float* calib_w = (const float*)d_in[25];
  const float* calib_b = (const float*)d_in[26];
  const float* rg_w1   = (const float*)d_in[27];
  const float* rg_w2   = (const float*)d_in[28];
  const float* rg_b2   = (const float*)d_in[29];
  const float* zetaP   = (const float*)d_in[30];
  const float* thetaP  = (const float*)d_in[31];
  const float* lambdaP = (const float*)d_in[32];
  float* outF = (float*)d_out;

  const int E[4] = {2048, 4096, 6144, 8192};

  // Workspace bump allocator (~600 MB total).
  char* wp = (char*)d_ws;
  auto alloc = [&](size_t bytes) -> char* {
    char* q = wp;
    wp += (bytes + 255) & ~(size_t)255;
    return q;
  };
  __bf16* xb   = (__bf16*)alloc((size_t)BATCH * IN_DIM * 2);
  __bf16* hb   = (__bf16*)alloc((size_t)BATCH * IN_DIM * 2);
  __bf16* shb  = (__bf16*)alloc((size_t)BATCH * IN_DIM * 2);
  __bf16* Wb   = (__bf16*)alloc((size_t)OUT_PAD * IN_DIM * 2);
  __bf16* upb[4], *dnb[4], *domb_[4], *mid[4];
  for (int j = 0; j < 4; ++j) upb[j]   = (__bf16*)alloc((size_t)E[j] * IN_DIM * 2);
  for (int j = 0; j < 4; ++j) dnb[j]   = (__bf16*)alloc((size_t)OUT_PAD * E[j] * 2);
  for (int k = 0; k < 4; ++k) domb_[k] = (__bf16*)alloc((size_t)OUT_PAD * IN_DIM * 2);
  __bf16* calb  = (__bf16*)alloc((size_t)OUT_PAD * IN_DIM * 2);
  __bf16* rg1b  = (__bf16*)alloc((size_t)RG_DIM * IN_DIM * 2);
  __bf16* rg2b  = (__bf16*)alloc((size_t)OUT_PAD * RG_DIM * 2);
  for (int j = 0; j < 4; ++j) mid[j]   = (__bf16*)alloc((size_t)BATCH * E[j] * 2);
  __bf16* rgmid = (__bf16*)alloc((size_t)BATCH * RG_DIM * 2);
  float* cCoef  = (float*)alloc((size_t)4 * BATCH * 4);
  float* dCoef  = (float*)alloc((size_t)4 * BATCH * 4);

  auto conv = [&](const float* s, __bf16* d, long long sr, long long cols, long long dr) {
    long long tot = dr * cols;
    int blocks = (int)((tot + 255) / 256);
    convert_pad_kernel<<<blocks, 256, 0, stream>>>(s, d, sr, cols, dr);
  };

  // 1) bf16 conversion of all weights (pad N=1000 -> 1024 with zero rows).
  conv(W, Wb, OUT_DIM, IN_DIM, OUT_PAD);
  for (int j = 0; j < 4; ++j) conv(up_w[j], upb[j], E[j], IN_DIM, E[j]);
  for (int j = 0; j < 4; ++j) conv(down_w[j], dnb[j], OUT_DIM, E[j], OUT_PAD);
  for (int k = 0; k < 4; ++k)
    conv(dom_ew + (size_t)k * OUT_DIM * IN_DIM, domb_[k], OUT_DIM, IN_DIM, OUT_PAD);
  conv(calib_w, calb, OUT_DIM, IN_DIM, OUT_PAD);
  conv(rg_w1, rg1b, RG_DIM, IN_DIM, RG_DIM);
  conv(rg_w2, rg2b, OUT_DIM, RG_DIM, OUT_PAD);

  // 2) LayerNorm + activation feature maps in bf16.
  prep_x_kernel<<<BATCH, 256, 0, stream>>>(x, ln_g, ln_b, xb, hb, shb);

  // 3) Router coefficients (softmaxes folded into per-row scalars).
  router_kernel<<<BATCH, 256, 0, stream>>>(x, hb,
      r_w[0], r_b[0], r_w[1], r_b[1], r_w[2], r_b[2], r_w[3], r_b[3],
      dom_rw, dom_rb, scales, zetaP, cCoef, dCoef);

  const dim3 blk(256);
  const dim3 gOut(BATCH / 128, OUT_PAD / 256);

  // 4) base_logits -> initializes d_out.
  gemm_bf16_wmma<0, 0><<<gOut, blk, 0, stream>>>(xb, IN_DIM, Wb, IN_DIM, IN_DIM,
      outF, nullptr, OUT_DIM, OUT_DIM, bv, nullptr, nullptr);

  // 5) Expert up-projections with fused activations -> bf16 mids.
  gemm_bf16_wmma<3, 0><<<dim3(BATCH / 128, E[0] / 256), blk, 0, stream>>>(
      xb, IN_DIM, upb[0], IN_DIM, IN_DIM, nullptr, mid[0], E[0], E[0], nullptr, nullptr, nullptr);
  gemm_bf16_wmma<3, 1><<<dim3(BATCH / 128, E[1] / 256), blk, 0, stream>>>(
      xb, IN_DIM, upb[1], IN_DIM, IN_DIM, nullptr, mid[1], E[1], E[1], nullptr, nullptr, nullptr);
  gemm_bf16_wmma<3, 2><<<dim3(BATCH / 128, E[2] / 256), blk, 0, stream>>>(
      xb, IN_DIM, upb[2], IN_DIM, IN_DIM, nullptr, mid[2], E[2], E[2], nullptr, nullptr, nullptr);
  gemm_bf16_wmma<3, 3><<<dim3(BATCH / 128, E[3] / 256), blk, 0, stream>>>(
      xb, IN_DIM, upb[3], IN_DIM, IN_DIM, nullptr, mid[3], E[3], E[3], nullptr, nullptr, nullptr);

  // 6) Reasoning stage 1 (silu) -> rgmid.
  gemm_bf16_wmma<3, 0><<<dim3(BATCH / 128, RG_DIM / 256), blk, 0, stream>>>(
      xb, IN_DIM, rg1b, IN_DIM, IN_DIM, nullptr, rgmid, RG_DIM, RG_DIM, nullptr, nullptr, nullptr);

  // 7) Down-projections, accumulated with per-row folded routing coefficients.
  for (int j = 0; j < 4; ++j)
    gemm_bf16_wmma<1, 0><<<gOut, blk, 0, stream>>>(mid[j], E[j], dnb[j], E[j], E[j],
        outF, nullptr, OUT_DIM, OUT_DIM, nullptr, cCoef + (size_t)j * BATCH, nullptr);

  // 8) Domain experts, accumulated with zeta * dom softmax weights per row.
  for (int k = 0; k < 4; ++k)
    gemm_bf16_wmma<1, 0><<<gOut, blk, 0, stream>>>(shb, IN_DIM, domb_[k], IN_DIM, IN_DIM,
        outF, nullptr, OUT_DIM, OUT_DIM, nullptr, dCoef + (size_t)k * BATCH, nullptr);

  // 9) calib: += theta * tanh(h @ calib_w.T + calib_b)
  gemm_bf16_wmma<2, 0><<<gOut, blk, 0, stream>>>(hb, IN_DIM, calb, IN_DIM, IN_DIM,
      outF, nullptr, OUT_DIM, OUT_DIM, calib_b, nullptr, thetaP);

  // 10) reason: += lambda * tanh(rgmid @ rg_w2.T + rg_b2)
  gemm_bf16_wmma<2, 0><<<gOut, blk, 0, stream>>>(rgmid, RG_DIM, rg2b, RG_DIM, RG_DIM,
      outF, nullptr, OUT_DIM, OUT_DIM, rg_b2, nullptr, lambdaP);

  (void)in_sizes; (void)n_in; (void)out_size; (void)ws_size;
}
